// LSTM_73512660238908
// MI455X (gfx1250) — 8-bit matrix operand order repaired and checked by reading, compiled, not run on hardware
//
#include <hip/hip_runtime.h>

#define Bq   128
#define Tq   1024
#define Hq   256
#define Eq   64
#define Cq   10

typedef __attribute__((ext_vector_type(16))) int   v16i;
typedef __attribute__((ext_vector_type(8)))  float v8f;
typedef __attribute__((ext_vector_type(4)))  int   intv4;

union V16 { v16i v; intv4 q[4]; };

__device__ __forceinline__ unsigned char f32_to_fp8_sw(float x) {
    unsigned int u  = __float_as_uint(x);
    unsigned char s = (unsigned char)((u >> 24) & 0x80u);
    unsigned int ax = u & 0x7fffffffu;
    if (ax >= 0x43e00000u) return (unsigned char)(s | 0x7Eu);
    if (ax <  0x3a800000u) return s;
    unsigned int e = ax >> 23;
    unsigned int m = ax & 0x7fffffu;
    unsigned int r = m + 0x7ffffu + ((m >> 20) & 1u);
    if (r & 0x800000u) { e += 1u; r = 0u; }
    if (e < 121u) return s;
    unsigned char E = (unsigned char)(e - 120u);
    if (E > 15u) return (unsigned char)(s | 0x7Eu);
    return (unsigned char)(s | (E << 3) | (unsigned char)((r >> 20) & 7u));
}

__device__ __forceinline__ unsigned char f32_to_fp8(float x) {
#if __has_builtin(__builtin_amdgcn_cvt_pk_fp8_f32)
    return (unsigned char)(__builtin_amdgcn_cvt_pk_fp8_f32(x, x, 0, false) & 0xff);
#else
    return f32_to_fp8_sw(x);
#endif
}

__device__ __forceinline__ float fast_sigmoid(float x) {
    return __builtin_amdgcn_rcpf(1.0f + __builtin_amdgcn_exp2f(-1.44269504f * x));
}
__device__ __forceinline__ float fast_tanh(float x) {
    return 1.0f - 2.0f * __builtin_amdgcn_rcpf(__builtin_amdgcn_exp2f(2.88539008f * x) + 1.0f);
}

__global__ void pack_wh_fp8(const float* __restrict__ Wf, const float* __restrict__ Wi,
                            const float* __restrict__ Wg, const float* __restrict__ Wo,
                            unsigned char* __restrict__ wb) {
    int gid = blockIdx.x * blockDim.x + threadIdx.x;
    if (gid >= 4 * 16 * 2 * 2048) return;
    int frag = gid >> 11;
    int rem  = gid & 2047;
    int lane = rem >> 6;
    int qq   = rem & 63;
    int vg   = qq >> 2;
    int byt  = qq & 3;
    int kf = frag & 1;
    int w  = (frag >> 1) & 15;
    int g  = frag >> 5;
    int N     = lane & 15;
    int halfK = lane >> 4;
    int grp   = vg >> 2;
    int vsub  = vg & 3;
    int K   = kf * 128 + grp * 32 + halfK * 16 + vsub * 4 + byt;
    int col = w * 16 + N;
    const float* W = (g == 0) ? Wf : (g == 1) ? Wi : (g == 2) ? Wg : Wo;
    wb[gid] = f32_to_fp8_sw(W[K * Hq + col]);
}

__global__ void pack_exg(const float* __restrict__ emb,
                         const float* __restrict__ Wfx, const float* __restrict__ Wix,
                         const float* __restrict__ Wgx, const float* __restrict__ Wox,
                         const float* __restrict__ bf,  const float* __restrict__ bi,
                         const float* __restrict__ bg,  const float* __restrict__ bo,
                         float* __restrict__ exg) {
    int gid = blockIdx.x * blockDim.x + threadIdx.x;
    if (gid >= 3 * 1024) return;
    int v   = gid >> 10;
    int n   = gid & 1023;
    int g   = n >> 8;
    int col = n & 255;
    const float* Wx = (g == 0) ? Wfx : (g == 1) ? Wix : (g == 2) ? Wgx : Wox;
    const float* bb = (g == 0) ? bf  : (g == 1) ? bi  : (g == 2) ? bg  : bo;
    float s = bb[col];
    for (int k = 0; k < Eq; ++k) s += emb[v * Eq + k] * Wx[k * Hq + col];
    exg[gid] = s;
}

__global__ __launch_bounds__(512, 1)
void lstm_fp8(const int* __restrict__ x,
              const unsigned char* __restrict__ wb,
              const float* __restrict__ exg,
              const float* __restrict__ Wph, const float* __restrict__ Wpb,
              float* __restrict__ out) {
    __shared__ unsigned char hA[2][4096];
    __shared__ unsigned int  ptok[Tq];
    __shared__ float hf[16 * 256];
    __shared__ float pbuf[16 * Cq];

    const int tid  = threadIdx.x;
    const int lane = tid & 31;
    const int w    = tid >> 5;
    const int bt   = blockIdx.x * 16;

    for (int t = tid; t < Tq; t += 512) {
        unsigned int wv = 0;
#pragma unroll
        for (int m = 0; m < 16; ++m)
            wv |= ((unsigned int)x[(bt + m) * Tq + t] & 3u) << (2 * m);
        ptok[t] = wv;
    }
    for (int idx = tid; idx < 2 * 4096; idx += 512) ((unsigned char*)hA)[idx] = 0;

#pragma unroll
    for (int g = 0; g < 4; ++g)
#pragma unroll
        for (int kf = 0; kf < 2; ++kf)
            __builtin_prefetch(wb + (((g * 16 + w) * 2 + kf) * 2048) + lane * 64, 0, 3);

    V16 Wb[4][2];
#pragma unroll
    for (int g = 0; g < 4; ++g)
#pragma unroll
        for (int kf = 0; kf < 2; ++kf) {
            const intv4* p =
                (const intv4*)(wb + (((g * 16 + w) * 2 + kf) * 2048) + lane * 64);
            Wb[g][kf].q[0] = p[0];
            Wb[g][kf].q[1] = p[1];
            Wb[g][kf].q[2] = p[2];
            Wb[g][kf].q[3] = p[3];
        }

    const int jcol = w * 16 + (lane & 15);
    const int mo   = (lane >> 4) * 8;

    float xv[4][3];
#pragma unroll
    for (int g = 0; g < 4; ++g)
#pragma unroll
        for (int v = 0; v < 3; ++v) xv[g][v] = exg[v * 1024 + g * 256 + jcol];

    int a8[8];
    {
        int kf    = jcol >> 7;
        int Kp    = jcol & 127;
        int halfA = (Kp >> 4) & 1;
        int vg    = ((Kp >> 5) & 3) * 4 + ((Kp >> 2) & 3);
        int byt   = Kp & 3;
#pragma unroll
        for (int r = 0; r < 8; ++r) {
            int lp = (r + mo) + 16 * halfA;
            a8[r]  = kf * 2048 + lp * 64 + vg * 4 + byt;
        }
    }

    __syncthreads();

    v8f cst = {0.f, 0.f, 0.f, 0.f, 0.f, 0.f, 0.f, 0.f};
    v8f hv  = {0.f, 0.f, 0.f, 0.f, 0.f, 0.f, 0.f, 0.f};

    for (int t = 0; t < Tq; ++t) {
        const int cur = t & 1, nxt = cur ^ 1;

        V16 A0, A1;
        {
            const intv4* ap0 = (const intv4*)(&hA[cur][lane * 64]);
            const intv4* ap1 = (const intv4*)(&hA[cur][2048 + lane * 64]);
            A0.q[0] = ap0[0]; A0.q[1] = ap0[1]; A0.q[2] = ap0[2]; A0.q[3] = ap0[3];
            A1.q[0] = ap1[0]; A1.q[1] = ap1[1]; A1.q[2] = ap1[2]; A1.q[3] = ap1[3];
        }

        const unsigned int tw = ptok[t];
        v8f accf, acci, accg, acco;
#pragma unroll
        for (int r = 0; r < 8; ++r) {
            const int tk = (int)((tw >> (2 * (r + mo))) & 3u);
            accf[r] = (tk == 0) ? xv[0][0] : (tk == 1) ? xv[0][1] : xv[0][2];
            acci[r] = (tk == 0) ? xv[1][0] : (tk == 1) ? xv[1][1] : xv[1][2];
            accg[r] = (tk == 0) ? xv[2][0] : (tk == 1) ? xv[2][1] : xv[2][2];
            acco[r] = (tk == 0) ? xv[3][0] : (tk == 1) ? xv[3][1] : xv[3][2];
        }

        accf = __builtin_amdgcn_wmma_f32_16x16x128_fp8_fp8(A0.v, Wb[0][0].v, (short)0, accf, false, false);
        accf = __builtin_amdgcn_wmma_f32_16x16x128_fp8_fp8(A1.v, Wb[0][1].v, (short)0, accf, false, false);
        acci = __builtin_amdgcn_wmma_f32_16x16x128_fp8_fp8(A0.v, Wb[1][0].v, (short)0, acci, false, false);
        acci = __builtin_amdgcn_wmma_f32_16x16x128_fp8_fp8(A1.v, Wb[1][1].v, (short)0, acci, false, false);
        accg = __builtin_amdgcn_wmma_f32_16x16x128_fp8_fp8(A0.v, Wb[2][0].v, (short)0, accg, false, false);
        accg = __builtin_amdgcn_wmma_f32_16x16x128_fp8_fp8(A1.v, Wb[2][1].v, (short)0, accg, false, false);
        acco = __builtin_amdgcn_wmma_f32_16x16x128_fp8_fp8(A0.v, Wb[3][0].v, (short)0, acco, false, false);
        acco = __builtin_amdgcn_wmma_f32_16x16x128_fp8_fp8(A1.v, Wb[3][1].v, (short)0, acco, false, false);

#pragma unroll
        for (int r = 0; r < 8; ++r) {
            float fr  = fast_sigmoid(accf[r]);
            float ir  = fast_sigmoid(acci[r]);
            float gr  = fast_tanh   (accg[r]);
            float orv = fast_sigmoid(acco[r]);
            float cr  = fr * cst[r] + ir * gr;
            cst[r]    = cr;
            float hr  = fast_tanh(cr) * orv;
            hv[r]     = hr;
            hA[nxt][a8[r]] = f32_to_fp8(hr);
        }
        __syncthreads();
    }

#pragma unroll
    for (int r = 0; r < 8; ++r) hf[(r + mo) * 256 + jcol] = hv[r];
    __syncthreads();

    if (tid < 16 * Cq) {
        int b = tid / Cq, cls = tid % Cq;
        float s = Wpb[cls];
        for (int k = 0; k < Hq; ++k) s += hf[b * 256 + k] * Wph[k * Cq + cls];
        pbuf[tid] = s;
    }
    __syncthreads();

    if (tid < 16) {
        float row[Cq];
        float m = -1e30f;
#pragma unroll
        for (int c = 0; c < Cq; ++c) { row[c] = pbuf[tid * Cq + c]; m = fmaxf(m, row[c]); }
        float s = 0.f;
#pragma unroll
        for (int c = 0; c < Cq; ++c) s += __expf(row[c] - m);
        float ls = __logf(s);
#pragma unroll
        for (int c = 0; c < Cq; ++c) out[(bt + tid) * Cq + c] = row[c] - m - ls;
    }
}

extern "C" void kernel_launch(void* const* d_in, const int* in_sizes, int n_in,
                              void* d_out, int out_size, void* d_ws, size_t ws_size,
                              hipStream_t stream) {
    (void)in_sizes; (void)n_in; (void)out_size; (void)ws_size;
    const int*   x   = (const int*)  d_in[0];
    const float* emb = (const float*)d_in[1];
    const float* Wfx = (const float*)d_in[2];
    const float* Wfh = (const float*)d_in[3];
    const float* Wfb = (const float*)d_in[4];
    const float* Wix = (const float*)d_in[5];
    const float* Wih = (const float*)d_in[6];
    const float* Wib = (const float*)d_in[7];
    const float* Wgx = (const float*)d_in[8];
    const float* Wgh = (const float*)d_in[9];
    const float* Wgb = (const float*)d_in[10];
    const float* Wox = (const float*)d_in[11];
    const float* Woh = (const float*)d_in[12];
    const float* Wob = (const float*)d_in[13];
    const float* Wph = (const float*)d_in[14];
    const float* Wpb = (const float*)d_in[15];

    unsigned char* wb  = (unsigned char*)d_ws;
    float*         exg = (float*)((char*)d_ws + 262144);

    pack_wh_fp8<<<1024, 256, 0, stream>>>(Wfh, Wih, Wgh, Woh, wb);
    pack_exg<<<12, 256, 0, stream>>>(emb, Wfx, Wix, Wgx, Wox, Wfb, Wib, Wgb, Wob, exg);
    lstm_fp8<<<8, 512, 0, stream>>>(x, wb, exg, Wph, Wpb, (float*)d_out);
}
